// SelfAttention_5360119185414
// MI455X (gfx1250) — matrix operand order repaired and checked by reading, not compiled, not run on hardware
//
#include <hip/hip_runtime.h>
#include <hip/hip_bf16.h>

typedef _Float16 f16;
typedef __attribute__((ext_vector_type(16))) _Float16 v16h;
typedef __attribute__((ext_vector_type(8)))  _Float16 v8h;
typedef __attribute__((ext_vector_type(8)))  float    v8f;

static constexpr int Bb = 2;
static constexpr int S  = 2048;
static constexpr int D  = 1024;
static constexpr int H  = 16;
static constexpr int HD = 64;
static constexpr int M  = Bb * S;   // 4096 rows

__device__ __forceinline__ v8f wmma_f16(v16h a, v16h b, v8f c) {
  return __builtin_amdgcn_wmma_f32_16x16x32_f16(false, a, false, b, (short)0, c,
                                                false, false);
}

// Low 32 bits of a flat pointer into __shared__ == AS(3) byte offset from the
// workgroup LDS base, which is what async-to-LDS VDST expects.
__device__ __forceinline__ unsigned lds_off_u32(const void* p) {
  return (unsigned)(uintptr_t)p;
}

// One 16-byte async copy global -> LDS (per lane), tracked by ASYNCcnt.
__device__ __forceinline__ void async_cp16(unsigned lds_byte_off,
                                           const void* gsrc) {
  asm volatile("global_load_async_to_lds_b128 %0, %1, off"
               :: "v"(lds_byte_off), "v"(gsrc)
               : "memory");
}

// ---------------------------------------------------------------------------
// fp32 -> f16 downconvert
// ---------------------------------------------------------------------------
__global__ void cvt_f32_f16(const float* __restrict__ src, f16* __restrict__ dst,
                            int n) {
  int i = blockIdx.x * blockDim.x + threadIdx.x;
  if (i < n) dst[i] = (f16)src[i];
}

// ---------------------------------------------------------------------------
// Fused QKV projection with async-to-LDS double-buffered B tiles.
// X: [M, D] f16.  W: [D, D] f16 (out, in).  Q,K -> [B,H,S,HD]; V -> [B,H,HD,S].
// Block = 8 waves; wave computes 16(M) x 64(N); block tile 128 x 64.
// LDS: 2 buf x 3 mat x (64 x 32) f16 = 24 KB.
// ---------------------------------------------------------------------------
__global__ __launch_bounds__(256) void qkv_gemm(
    const f16* __restrict__ Xh,
    const f16* __restrict__ Wq, const f16* __restrict__ Wk,
    const f16* __restrict__ Wv,
    const float* __restrict__ bq, const float* __restrict__ bk,
    const float* __restrict__ bv,
    f16* __restrict__ Qo, f16* __restrict__ Ko, f16* __restrict__ Vt) {
  __shared__ f16 ldsB[2][3][64][32];

  const int tid  = threadIdx.x;
  const int lane = tid & 31;
  const int wave = tid >> 5;
  const int half = lane >> 4;
  const int l15  = lane & 15;

  const int mBase = blockIdx.x * 128 + wave * 16;
  const int nBase = blockIdx.y * 64;

  // Async-copy coordinates: thread tid copies f16 elements [tid*8, tid*8+8)
  // of the 64x32 tile:  n = tid>>2, k-in-tile = (tid&3)*8.
  const unsigned cpOff = (unsigned)(tid * 16);  // byte offset inside one tile
  const f16* gq = Wq + (unsigned)((nBase + (tid >> 2)) * D + ((tid & 3) << 3));
  const f16* gk = Wk + (unsigned)((nBase + (tid >> 2)) * D + ((tid & 3) << 3));
  const f16* gv = Wv + (unsigned)((nBase + (tid >> 2)) * D + ((tid & 3) << 3));
  const unsigned ldsBase = lds_off_u32(&ldsB[0][0][0][0]);

  const f16* arow = Xh + (unsigned)((mBase + l15) * D);

  v8f accQ[4] = {}, accK[4] = {}, accV[4] = {};

  auto issue_stage = [&](int buf, int k) {
    const unsigned b = ldsBase + (unsigned)buf * (3u * 4096u) + cpOff;
    async_cp16(b,           gq + k);
    async_cp16(b + 4096u,   gk + k);
    async_cp16(b + 8192u,   gv + k);
  };

  auto load_a = [&](int k) -> v16h {
    v8h a0 = *reinterpret_cast<const v8h*>(arow + k + 16 * half);
    v8h a1 = *reinterpret_cast<const v8h*>(arow + k + 16 * half + 8);
    v16h a;
#pragma unroll
    for (int i = 0; i < 8; ++i) { a[i] = a0[i]; a[i + 8] = a1[i]; }
    return a;
  };

  constexpr int KSTEPS = D / 32;  // 32
  issue_stage(0, 0);
  v16h aCur = load_a(0);

#pragma unroll 2
  for (int ks = 0; ks < KSTEPS; ++ks) {
    const int cur = ks & 1;
    v16h aNext = aCur;
    if (ks + 1 < KSTEPS) {
      issue_stage(cur ^ 1, (ks + 1) * 32);
      aNext = load_a((ks + 1) * 32);
      asm volatile("s_wait_asynccnt 0x3" ::: "memory");
    } else {
      asm volatile("s_wait_asynccnt 0x0" ::: "memory");
    }
    __syncthreads();

    // B-fragment LDS reads pipelined one t-step ahead so WMMAs overlap dscnt.
    auto ldb = [&](int mat, int t) -> v16h {
      return *reinterpret_cast<const v16h*>(
          &ldsB[cur][mat][t * 16 + l15][16 * half]);
    };
    v16h c0 = ldb(0, 0), c1 = ldb(1, 0), c2 = ldb(2, 0);
#pragma unroll
    for (int t = 0; t < 4; ++t) {
      v16h n0, n1, n2;
      if (t < 3) { n0 = ldb(0, t + 1); n1 = ldb(1, t + 1); n2 = ldb(2, t + 1); }
      accQ[t] = wmma_f16(aCur, c0, accQ[t]);
      accK[t] = wmma_f16(aCur, c1, accK[t]);
      accV[t] = wmma_f16(aCur, c2, accV[t]);
      if (t < 3) { c0 = n0; c1 = n1; c2 = n2; }
    }
    __syncthreads();
    aCur = aNext;
  }

  // Epilogue: C element (j, lane): row = j + 8*half, col = l15
#pragma unroll
  for (int t = 0; t < 4; ++t) {
    const int n = nBase + t * 16 + l15;
    const int h = n >> 6, hd = n & 63;
    const float vbq = bq[n], vbk = bk[n], vbv = bv[n];
#pragma unroll
    for (int j = 0; j < 8; ++j) {
      const int m = mBase + j + 8 * half;
      const int b = m >> 11, s = m & (S - 1);
      const unsigned bh = (unsigned)(b * H + h);
      Qo[(bh * S + s) * HD + hd] = (f16)(accQ[t][j] + vbq);
      Ko[(bh * S + s) * HD + hd] = (f16)(accK[t][j] + vbk);
      Vt[(bh * HD + hd) * S + s] = (f16)(accV[t][j] + vbv);
    }
  }
}

// ---------------------------------------------------------------------------
// Flash attention: one wave owns a 16-row Q tile of one (b, h).
// K fragments double-buffered in registers; V issued early each step.
// 1/sqrt(HD) is folded into the Q fragment.
// ---------------------------------------------------------------------------
__global__ __launch_bounds__(256) void attn_flash(
    const f16* __restrict__ Q, const f16* __restrict__ Kc,
    const f16* __restrict__ Vt, f16* __restrict__ Oh) {
  __shared__ f16 plds[8][16][32];  // per-wave P tile

  const int lane = threadIdx.x & 31;
  const int wave = threadIdx.x >> 5;
  const int half = lane >> 4;
  const int l15  = lane & 15;

  const int gw = blockIdx.x * 8 + wave;
  const int qt = gw & (S / 16 - 1);
  const int bh = gw / (S / 16);

  const f16* qrow  = Q  + (unsigned)((bh * S + qt * 16 + l15) * HD);
  const f16* kbase = Kc + (unsigned)(bh * S * HD);
  const f16* vbase = Vt + (unsigned)(bh * HD * S);

  const f16 qscale = (f16)0.125f;  // 1/sqrt(64) folded into Q
  v16h aq[2];
#pragma unroll
  for (int kk = 0; kk < 2; ++kk) {
    v8h a0 = *reinterpret_cast<const v8h*>(qrow + kk * 32 + 16 * half);
    v8h a1 = *reinterpret_cast<const v8h*>(qrow + kk * 32 + 16 * half + 8);
#pragma unroll
    for (int i = 0; i < 8; ++i) {
      aq[kk][i] = a0[i] * qscale;
      aq[kk][i + 8] = a1[i] * qscale;
    }
  }

  auto loadK = [&](int st, v16h kf[4]) {
#pragma unroll
    for (int kk = 0; kk < 2; ++kk) {
      const int dofs = kk * 32 + 16 * half;
      kf[kk * 2 + 0] = *reinterpret_cast<const v16h*>(
          kbase + (unsigned)((st + l15) * HD) + dofs);
      kf[kk * 2 + 1] = *reinterpret_cast<const v16h*>(
          kbase + (unsigned)((st + 16 + l15) * HD) + dofs);
    }
  };

  float mrun[8], lrun[8];
#pragma unroll
  for (int j = 0; j < 8; ++j) { mrun[j] = -3.0e38f; lrun[j] = 0.0f; }
  v8f o[4] = {};

  v16h kbuf[2][4];
  loadK(0, kbuf[0]);

#pragma unroll 2
  for (int st = 0; st < S; st += 32) {
    const int cur = (st >> 5) & 1;

    // Issue V loads now; consumed only after the softmax VALU chain.
    v16h vf[4];
#pragma unroll
    for (int t = 0; t < 4; ++t)
      vf[t] = *reinterpret_cast<const v16h*>(
          vbase + (unsigned)((t * 16 + l15) * S) + st + 16 * half);
    // Prefetch next K tile into the other register buffer.
    if (st + 32 < S) loadK(st + 32, kbuf[cur ^ 1]);

    v8f sc0 = {}, sc1 = {};
    sc0 = wmma_f16(aq[0], kbuf[cur][0], sc0);
    sc1 = wmma_f16(aq[0], kbuf[cur][1], sc1);
    sc0 = wmma_f16(aq[1], kbuf[cur][2], sc0);
    sc1 = wmma_f16(aq[1], kbuf[cur][3], sc1);

    float alpha[8];
#pragma unroll
    for (int j = 0; j < 8; ++j) {
      float x0 = sc0[j], x1 = sc1[j];
      float mx = fmaxf(x0, x1);
#pragma unroll
      for (int off = 1; off < 16; off <<= 1)
        mx = fmaxf(mx, __shfl_xor(mx, off, 32));
      const float mnew = fmaxf(mrun[j], mx);
      const float a = __expf(mrun[j] - mnew);
      const float p0 = __expf(x0 - mnew);
      const float p1 = __expf(x1 - mnew);
      float psum = p0 + p1;
#pragma unroll
      for (int off = 1; off < 16; off <<= 1)
        psum += __shfl_xor(psum, off, 32);
      lrun[j] = lrun[j] * a + psum;
      mrun[j] = mnew;
      alpha[j] = a;
      sc0[j] = p0;
      sc1[j] = p1;
    }

    // P: C layout -> A layout via per-wave LDS tile (intra-wave, DScnt-ordered)
    f16(*pt)[32] = plds[wave];
#pragma unroll
    for (int j = 0; j < 8; ++j) {
      const int row = j + 8 * half;
      pt[row][l15]      = (f16)sc0[j];
      pt[row][16 + l15] = (f16)sc1[j];
    }
    v16h ap;
    {
      v8h c0 = *reinterpret_cast<const v8h*>(&pt[l15][16 * half]);
      v8h c1 = *reinterpret_cast<const v8h*>(&pt[l15][16 * half + 8]);
#pragma unroll
      for (int i = 0; i < 8; ++i) { ap[i] = c0[i]; ap[i + 8] = c1[i]; }
    }

#pragma unroll
    for (int t = 0; t < 4; ++t) {
#pragma unroll
      for (int j = 0; j < 8; ++j) o[t][j] *= alpha[j];
      o[t] = wmma_f16(ap, vf[t], o[t]);
    }
  }

  const int b = bh >> 4, h = bh & 15;
  float inv[8];
#pragma unroll
  for (int j = 0; j < 8; ++j) inv[j] = 1.0f / lrun[j];
#pragma unroll
  for (int t = 0; t < 4; ++t) {
#pragma unroll
    for (int j = 0; j < 8; ++j) {
      const int sq = qt * 16 + j + 8 * half;
      const int dc = t * 16 + l15;
      Oh[(unsigned)((b * S + sq) * D + h * HD + dc)] = (f16)(o[t][j] * inv[j]);
    }
  }
}

// ---------------------------------------------------------------------------
// Output projection: out = Oh @ Wo^T + bo (f32), async-to-LDS staged B tiles.
// ---------------------------------------------------------------------------
__global__ __launch_bounds__(256) void out_gemm(
    const f16* __restrict__ Ah, const f16* __restrict__ W,
    const float* __restrict__ bias, float* __restrict__ out) {
  __shared__ f16 ldsB[2][64][32];

  const int tid  = threadIdx.x;
  const int lane = tid & 31;
  const int wave = tid >> 5;
  const int half = lane >> 4;
  const int l15  = lane & 15;

  const int mBase = blockIdx.x * 128 + wave * 16;
  const int nBase = blockIdx.y * 64;

  const unsigned cpOff = (unsigned)(tid * 16);
  const f16* gw = W + (unsigned)((nBase + (tid >> 2)) * D + ((tid & 3) << 3));
  const unsigned ldsBase = lds_off_u32(&ldsB[0][0][0]);

  const f16* arow = Ah + (unsigned)((mBase + l15) * D);

  v8f acc[4] = {};

  auto load_a = [&](int k) -> v16h {
    v8h a0 = *reinterpret_cast<const v8h*>(arow + k + 16 * half);
    v8h a1 = *reinterpret_cast<const v8h*>(arow + k + 16 * half + 8);
    v16h a;
#pragma unroll
    for (int i = 0; i < 8; ++i) { a[i] = a0[i]; a[i + 8] = a1[i]; }
    return a;
  };

  constexpr int KSTEPS = D / 32;
  async_cp16(ldsBase + cpOff, gw);
  v16h aCur = load_a(0);

#pragma unroll 2
  for (int ks = 0; ks < KSTEPS; ++ks) {
    const int cur = ks & 1;
    v16h aNext = aCur;
    if (ks + 1 < KSTEPS) {
      async_cp16(ldsBase + (unsigned)(cur ^ 1) * 4096u + cpOff,
                 gw + (ks + 1) * 32);
      aNext = load_a((ks + 1) * 32);
      asm volatile("s_wait_asynccnt 0x1" ::: "memory");
    } else {
      asm volatile("s_wait_asynccnt 0x0" ::: "memory");
    }
    __syncthreads();

    auto ldb = [&](int t) -> v16h {
      return *reinterpret_cast<const v16h*>(
          &ldsB[cur][t * 16 + l15][16 * half]);
    };
    v16h c0 = ldb(0);
#pragma unroll
    for (int t = 0; t < 4; ++t) {
      v16h n0;
      if (t < 3) n0 = ldb(t + 1);
      acc[t] = wmma_f16(aCur, c0, acc[t]);
      if (t < 3) c0 = n0;
    }
    __syncthreads();
    aCur = aNext;
  }

#pragma unroll
  for (int t = 0; t < 4; ++t) {
    const int n = nBase + t * 16 + l15;
    const float bn = bias[n];
#pragma unroll
    for (int j = 0; j < 8; ++j) {
      const int m = mBase + j + 8 * half;
      out[(unsigned)(m * D + n)] = acc[t][j] + bn;
    }
  }
}

// ---------------------------------------------------------------------------
extern "C" void kernel_launch(void* const* d_in, const int* in_sizes, int n_in,
                              void* d_out, int out_size, void* d_ws,
                              size_t ws_size, hipStream_t stream) {
  (void)in_sizes; (void)n_in; (void)out_size; (void)ws_size;
  const float* x  = (const float*)d_in[0];
  const float* Wq = (const float*)d_in[1];
  const float* bq = (const float*)d_in[2];
  const float* Wk = (const float*)d_in[3];
  const float* bk = (const float*)d_in[4];
  const float* Wv = (const float*)d_in[5];
  const float* bv = (const float*)d_in[6];
  const float* Wo = (const float*)d_in[7];
  const float* bo = (const float*)d_in[8];
  float* out = (float*)d_out;

  f16* Xh  = (f16*)d_ws;                 // M*D
  f16* Wqh = Xh  + (size_t)M * D;        // D*D
  f16* Wkh = Wqh + (size_t)D * D;
  f16* Wvh = Wkh + (size_t)D * D;
  f16* Woh = Wvh + (size_t)D * D;
  f16* Qb  = Woh + (size_t)D * D;        // [B,H,S,HD]
  f16* Kb  = Qb  + (size_t)M * D;        // [B,H,S,HD]
  f16* Vtb = Kb  + (size_t)M * D;        // [B,H,HD,S]
  f16* Ohb = Vtb + (size_t)M * D;        // [B,S,D]

  const int T = 256;
  cvt_f32_f16<<<(M * D) / T, T, 0, stream>>>(x, Xh, M * D);
  cvt_f32_f16<<<(D * D) / T, T, 0, stream>>>(Wq, Wqh, D * D);
  cvt_f32_f16<<<(D * D) / T, T, 0, stream>>>(Wk, Wkh, D * D);
  cvt_f32_f16<<<(D * D) / T, T, 0, stream>>>(Wv, Wvh, D * D);
  cvt_f32_f16<<<(D * D) / T, T, 0, stream>>>(Wo, Woh, D * D);

  dim3 gGemm(M / 128, D / 64);
  qkv_gemm<<<gGemm, T, 0, stream>>>(Xh, Wqh, Wkh, Wvh, bq, bk, bv, Qb, Kb, Vtb);

  const int waveTasks = Bb * H * (S / 16);  // 4096
  attn_flash<<<waveTasks / 8, T, 0, stream>>>(Qb, Kb, Vtb, Ohb);

  out_gemm<<<gGemm, T, 0, stream>>>(Ohb, Woh, bo, out);
}
